// HyperEncoder_87299505259186
// MI455X (gfx1250) — matrix operand order repaired and checked by reading, not compiled, not run on hardware
//
#include <hip/hip_runtime.h>
#include <hip/hip_bf16.h>

typedef __attribute__((ext_vector_type(16))) __bf16 v16bf;
typedef __attribute__((ext_vector_type(8)))  __bf16 v8bf;
typedef __attribute__((ext_vector_type(8)))  float  v8f;

#define DIM 256

// ---------------------------------------------------------------------------
// Pack a 256x256 fp32 weight matrix into bf16, ordered as this file feeds the CDNA5
// WMMA B-fragment (32x16 bf16), contiguous K per lane half (same as A):
//   flat = (((kt*16 + sub)*32 + lane)*16 + elem)
//   lane 0..15  -> col = sub*16 + lane,      K = kt*32 + 0..15  (elem)
//   lane 16..31 -> col = sub*16 + (lane-16), K = kt*32 + 16..31 (elem)
// One thread produces one lane's 16 contiguous bf16 (32 B store).
// ---------------------------------------------------------------------------
__global__ __launch_bounds__(256)
void pack_w_kernel(const float* __restrict__ W, __bf16* __restrict__ P)
{
    const int t    = blockIdx.x * 256 + threadIdx.x;   // 0..4095
    const int lane = t & 31;
    const int sub  = (t >> 5) & 15;
    const int kt   = t >> 9;                           // 0..7
    const int col   = sub * 16 + (lane & 15);
    const int kbase = kt * 32 + (lane >> 4) * 16;
    v16bf v;
    #pragma unroll
    for (int e = 0; e < 16; ++e)
        v[e] = (__bf16)W[(size_t)(kbase + e) * DIM + col];
    *(v16bf*)(P + (size_t)t * 16) = v;
}

// ---------------------------------------------------------------------------
// Tiled GEMM  C[M,256] = A[M,256] @ W[256,256]  (fp32 in/out, bf16 WMMA core)
// 256 threads = 8 waves; block tile 64x128; wave tile 16x64 (4 accumulators).
// A staged fp32->bf16 in LDS (B128 in/out); B fragments come pre-swizzled
// from global (one v16bf = 32 B per fragment per lane).
// ---------------------------------------------------------------------------
__global__ __launch_bounds__(256)
void gemm_wmma_bf16(const float* __restrict__ A, const __bf16* __restrict__ Wp,
                    float* __restrict__ C, int M)
{
    __shared__ __bf16 As[64][40];    // 64 rows x 32 K (+pad, rows 80 B apart)

    const int tid   = threadIdx.x;
    const int lane  = tid & 31;
    const int wave  = tid >> 5;
    const int waveM = wave >> 1;     // 0..3
    const int waveN = wave & 1;      // 0..1
    const int r0    = blockIdx.x * 64;
    const int c0    = blockIdx.y * 128;

    const int aRow = tid >> 2;          // 0..63
    const int aCol = (tid & 3) * 8;     // 0,8,16,24

    const int m  = 16 * waveM + (lane & 15);
    const int kh = (lane >> 4) * 16;         // lanes 16-31 hold K+16..K+31
    const int subBase = (c0 >> 4) + 4 * waveN;
    const v16bf* __restrict__ Bp = (const v16bf*)Wp;

    v8f acc[4] = {};

    for (int kk = 0; kk < DIM; kk += 32) {
        // ---- stage A tile (fp32 -> bf16), one 16 B LDS store per thread ----
        {
            const int gr = r0 + aRow;
            float4 f0 = make_float4(0.f, 0.f, 0.f, 0.f), f1 = f0;
            if (gr < M) {
                const float* ap = A + (size_t)gr * DIM + kk + aCol;
                f0 = *(const float4*)(ap);
                f1 = *(const float4*)(ap + 4);
            }
            v8bf pk;
            pk[0] = (__bf16)f0.x; pk[1] = (__bf16)f0.y;
            pk[2] = (__bf16)f0.z; pk[3] = (__bf16)f0.w;
            pk[4] = (__bf16)f1.x; pk[5] = (__bf16)f1.y;
            pk[6] = (__bf16)f1.z; pk[7] = (__bf16)f1.w;
            *(v8bf*)&As[aRow][aCol] = pk;
        }
        __syncthreads();

        // ---- A fragment: two B128 LDS loads (contiguous K per lane half, same order as packed B) ----
        v8bf lo = *(const v8bf*)&As[m][kh];
        v8bf hi = *(const v8bf*)&As[m][kh + 8];
        v16bf af = __builtin_shufflevector(lo, hi,
                    0, 1, 2, 3, 4, 5, 6, 7, 8, 9, 10, 11, 12, 13, 14, 15);

        // ---- 4 x (B fragment straight from pre-swizzled global + WMMA) ----
        const v16bf* bp = Bp + ((size_t)((kk >> 5) * 16 + subBase) * 32 + lane);
        #pragma unroll
        for (int j = 0; j < 4; ++j) {
            v16bf bfv = bp[(size_t)j * 32];
            acc[j] = __builtin_amdgcn_wmma_f32_16x16x32_bf16(
                false, af, false, bfv, (short)0, acc[j], false, false);
        }
        __syncthreads();
    }

    // ---- store C (ISA C/D layout: VGPR v -> row 8*(lane>=16)+v) ----
    #pragma unroll
    for (int j = 0; j < 4; ++j) {
        const int col   = c0 + 64 * waveN + 16 * j + (lane & 15);
        const int rbase = r0 + 16 * waveM + 8 * (lane >> 4);
        #pragma unroll
        for (int v = 0; v < 8; ++v) {
            const int row = rbase + v;
            if (row < M) C[(size_t)row * DIM + col] = acc[j][v];
        }
    }
}

// ---------------------------------------------------------------------------
__global__ __launch_bounds__(256)
void degree_kernel(const int* __restrict__ ni, const int* __restrict__ ei,
                   float* __restrict__ Dn, float* __restrict__ De, int nnz)
{
    int i = blockIdx.x * 256 + threadIdx.x;
    if (i < nnz) {
        unsafeAtomicAdd(&Dn[ni[i]], 1.0f);
        unsafeAtomicAdd(&De[ei[i]], 1.0f);
    }
}

__global__ __launch_bounds__(256)
void invert_kernel(float* __restrict__ d, int n)
{
    int i = blockIdx.x * 256 + threadIdx.x;
    if (i < n) { float v = d[i]; d[i] = (v > 0.f) ? (1.f / v) : 0.f; }
}

__global__ __launch_bounds__(256)
void fill_kernel(float* __restrict__ p, size_t n)
{
    size_t i = (size_t)blockIdx.x * 256 + threadIdx.x;
    if (i < n) p[i] = 0.f;
}

// ---------------------------------------------------------------------------
// scatter: one wave per nnz; lane handles 8 floats (2x float4 loads,
// 8 global_atomic_add_f32). dst[dstIdx[i]] += invDeg[dstIdx[i]] * src[srcIdx[i]]
// ---------------------------------------------------------------------------
__global__ __launch_bounds__(256)
void scatter_kernel(const float* __restrict__ src, const int* __restrict__ srcIdx,
                    const int* __restrict__ dstIdx, const float* __restrict__ invDeg,
                    float* __restrict__ dst, int nnz)
{
    const int g = blockIdx.x * 8 + (threadIdx.x >> 5);
    if (g >= nnz) return;
    const int lane = threadIdx.x & 31;
    const int si = srcIdx[g];
    const int di = dstIdx[g];
    const float w = invDeg[di];
    const float* sp = src + (size_t)si * DIM + lane * 8;
    float*       dp = dst + (size_t)di * DIM + lane * 8;
    float4 a = *(const float4*)(sp);
    float4 b = *(const float4*)(sp + 4);
    unsafeAtomicAdd(dp + 0, w * a.x);
    unsafeAtomicAdd(dp + 1, w * a.y);
    unsafeAtomicAdd(dp + 2, w * a.z);
    unsafeAtomicAdd(dp + 3, w * a.w);
    unsafeAtomicAdd(dp + 4, w * b.x);
    unsafeAtomicAdd(dp + 5, w * b.y);
    unsafeAtomicAdd(dp + 6, w * b.z);
    unsafeAtomicAdd(dp + 7, w * b.w);
}

// ---------------------------------------------------------------------------
__global__ __launch_bounds__(256)
void bias_prelu_kernel(float* __restrict__ p, const float* __restrict__ bias,
                       const float* __restrict__ pa, size_t n)
{
    size_t i = (size_t)blockIdx.x * 256 + threadIdx.x;
    if (i < n) {
        const float a = *pa;
        float v = p[i] + bias[i & (DIM - 1)];
        p[i] = (v >= 0.f) ? v : a * v;
    }
}

// ---------------------------------------------------------------------------
extern "C" void kernel_launch(void* const* d_in, const int* in_sizes, int n_in,
                              void* d_out, int out_size, void* d_ws, size_t ws_size,
                              hipStream_t stream)
{
    (void)in_sizes; (void)n_in; (void)out_size; (void)ws_size;
    const int N = 50000, E = 20000, NNZ = 400000;
    const int WELEM = DIM * DIM;  // 65536 bf16 per packed matrix

    const float* x        = (const float*)d_in[0];
    const int*   node_idx = (const int*)d_in[1];
    const int*   edge_idx = node_idx + NNZ;
    const float* W1_n2e = (const float*)d_in[4];
    const float* b1_n2e = (const float*)d_in[5];
    const float* W1_e2n = (const float*)d_in[6];
    const float* b1_e2n = (const float*)d_in[7];
    const float* W2_n2e = (const float*)d_in[8];
    const float* b2_n2e = (const float*)d_in[9];
    const float* W2_e2n = (const float*)d_in[10];
    const float* b2_e2n = (const float*)d_in[11];
    const float* pa     = (const float*)d_in[12];
    float* out = (float*)d_out;

    // workspace layout
    float*  ws    = (float*)d_ws;
    float*  Dn    = ws;                      // [N] -> Dn_inv
    float*  De    = ws + N;                  // [E] -> De_inv
    __bf16* Wpck  = (__bf16*)(ws + 70016);   // 4 x 65536 bf16 (512 KB)
    __bf16* Wp1a  = Wpck;
    __bf16* Wp1b  = Wpck + WELEM;
    __bf16* Wp2a  = Wpck + 2 * WELEM;
    __bf16* Wp2b  = Wpck + 3 * WELEM;
    float*  bufH  = ws + 70016 + 2 * WELEM;  // [N*256] (4*WELEM bf16 = 2*WELEM f32)
    float*  bufE  = bufH + (size_t)N * DIM;  // [E*256]
    float*  bufG  = bufE + (size_t)E * DIM;  // [E*256]

    const dim3 blk(256);
    const dim3 gemmN((N + 63) / 64, 2);
    const dim3 gemmE((E + 63) / 64, 2);
    const int scatBlocks = (NNZ + 7) / 8;

    // ---- one-time weight pre-swizzle (bf16, WMMA B-fragment order) ----
    pack_w_kernel<<<16, blk, 0, stream>>>(W1_n2e, Wp1a);
    pack_w_kernel<<<16, blk, 0, stream>>>(W1_e2n, Wp1b);
    pack_w_kernel<<<16, blk, 0, stream>>>(W2_n2e, Wp2a);
    pack_w_kernel<<<16, blk, 0, stream>>>(W2_e2n, Wp2b);

    // ---- degrees ----
    fill_kernel<<<(N + E + 255) / 256, blk, 0, stream>>>(Dn, (size_t)(N + E));
    degree_kernel<<<(NNZ + 255) / 256, blk, 0, stream>>>(node_idx, edge_idx, Dn, De, NNZ);
    invert_kernel<<<(N + E + 255) / 256, blk, 0, stream>>>(Dn, N + E);

    // ================= Layer 1 =================
    gemm_wmma_bf16<<<gemmN, blk, 0, stream>>>(x, Wp1a, bufH, N);
    fill_kernel<<<(size_t)E * DIM / 256, blk, 0, stream>>>(bufE, (size_t)E * DIM);
    scatter_kernel<<<scatBlocks, blk, 0, stream>>>(bufH, node_idx, edge_idx, De, bufE, NNZ);
    bias_prelu_kernel<<<(size_t)E * DIM / 256, blk, 0, stream>>>(bufE, b1_n2e, pa, (size_t)E * DIM);
    gemm_wmma_bf16<<<gemmE, blk, 0, stream>>>(bufE, Wp1b, bufG, E);
    fill_kernel<<<(size_t)N * DIM / 256, blk, 0, stream>>>(out, (size_t)N * DIM);
    scatter_kernel<<<scatBlocks, blk, 0, stream>>>(bufG, edge_idx, node_idx, Dn, out, NNZ);
    bias_prelu_kernel<<<(size_t)N * DIM / 256, blk, 0, stream>>>(out, b1_e2n, pa, (size_t)N * DIM);

    // ================= Layer 2 ================= (out currently holds h1)
    gemm_wmma_bf16<<<gemmN, blk, 0, stream>>>(out, Wp2a, bufH, N);
    fill_kernel<<<(size_t)E * DIM / 256, blk, 0, stream>>>(bufE, (size_t)E * DIM);
    scatter_kernel<<<scatBlocks, blk, 0, stream>>>(bufH, node_idx, edge_idx, De, bufE, NNZ);
    bias_prelu_kernel<<<(size_t)E * DIM / 256, blk, 0, stream>>>(bufE, b2_n2e, pa, (size_t)E * DIM);
    gemm_wmma_bf16<<<gemmE, blk, 0, stream>>>(bufE, Wp2b, bufG, E);
    fill_kernel<<<(size_t)N * DIM / 256, blk, 0, stream>>>(out, (size_t)N * DIM);
    scatter_kernel<<<scatBlocks, blk, 0, stream>>>(bufG, edge_idx, node_idx, Dn, out, NNZ);
    bias_prelu_kernel<<<(size_t)N * DIM / 256, blk, 0, stream>>>(out, b2_e2n, pa, (size_t)N * DIM);
}
